// AttentionModel_39848706572882
// MI455X (gfx1250) — matrix operand order repaired and checked by reading, not compiled, not run on hardware
//
#include <hip/hip_runtime.h>
#include <stdint.h>

#define Bx 2
#define Sx 2048
#define Hx 16
#define Ex 128

typedef __attribute__((ext_vector_type(16))) _Float16 v16h;
typedef __attribute__((ext_vector_type(8)))  _Float16 v8h;
typedef __attribute__((ext_vector_type(8)))  float    v8f;
typedef __attribute__((ext_vector_type(4)))  unsigned int v4u;
typedef __attribute__((ext_vector_type(8)))  int          v8i;
typedef __attribute__((ext_vector_type(4)))  int          v4i;

__device__ __forceinline__ v8f wmma_f16(v16h a, v16h b, v8f c) {
    // D = A(16x32 f16) * B(32x16 f16) + C(16x16 f32)
    return __builtin_amdgcn_wmma_f32_16x16x32_f16(
        false, a, false, b, (short)0, c, false, false);
}

__device__ __forceinline__ v8h pack8(float4 a, float4 b) {
    v8h r;
    r[0] = (_Float16)a.x; r[1] = (_Float16)a.y; r[2] = (_Float16)a.z; r[3] = (_Float16)a.w;
    r[4] = (_Float16)b.x; r[5] = (_Float16)b.y; r[6] = (_Float16)b.z; r[7] = (_Float16)b.w;
    return r;
}

#if defined(__gfx1250__) && __has_builtin(__builtin_amdgcn_tensor_load_to_lds) && \
    __has_builtin(__builtin_amdgcn_s_wait_tensorcnt)
#define USE_TDM 1
#endif

#ifdef USE_TDM
// Build a 2D Tensor-DMA descriptor (D#) and issue TENSOR_LOAD_TO_LDS.
// Layout per CDNA5 ISA:
//  g0: [1:0]=count=1 | lds_addr | global_addr[56:0] | type=2
//  g1: wg_mask=0, data_size=1(2B), tensor_dim0/1, tile_dim0/1 (tile_dim2=0 -> 2D),
//      tensor_dim0_stride; dim1_stride=0.  Groups 2/3 zero (unused for 2D).
// This toolchain exposes the 6-arg builtin:
//  (v4u g0, v8i g1, v4i g2, v4i g3, v8i g4, i32 cpol)
__device__ __forceinline__ void tdm_load_2d(unsigned lds_off, const void* gptr,
                                            unsigned td0, unsigned td1,
                                            unsigned tl0, unsigned tl1,
                                            unsigned stride0)
{
    unsigned long long ga = (unsigned long long)(uintptr_t)gptr;
    v4u g0;
    g0[0] = 1u;                                             // count=1, no gather
    g0[1] = lds_off;                                        // lds byte address
    g0[2] = (unsigned)ga;                                   // global_addr[31:0]
    g0[3] = (unsigned)((ga >> 32) & 0x1FFFFFFull) | (2u << 30); // addr[56:32] | type=2
    v8i g1;
    g1[0] = 0x00010000;                                     // data_size=1 (2 bytes)
    g1[1] = (int)((td0 & 0xFFFFu) << 16);                   // tensor_dim0[15:0]
    g1[2] = (int)(((td0 >> 16) & 0xFFFFu) | ((td1 & 0xFFFFu) << 16));
    g1[3] = (int)(((td1 >> 16) & 0xFFFFu) | ((tl0 & 0xFFFFu) << 16)); // tile_dim0
    g1[4] = (int)(tl1 & 0xFFFFu);                           // tile_dim1, tile_dim2=0
    g1[5] = (int)stride0;                                   // tensor_dim0_stride[31:0]
    g1[6] = 0;                                              // stride hi | dim1_stride lo
    g1[7] = 0;
    v4i z4 = {0, 0, 0, 0};
    v8i z8;
#pragma unroll
    for (int i = 0; i < 8; ++i) z8[i] = 0;
    __builtin_amdgcn_tensor_load_to_lds(g0, g1, z4, z4, z8, 0);
}
#endif

// ---------------------------------------------------------------------------
// Kernel 1: fused QKV projection.  Tiles are (b, h, 16 consecutive s) so all
// three output layouts get contiguous stores.  W is converted to f16 once per
// block into LDS; B-fragments come from LDS.  Q,K f16 [B,H,S,E]; V f16 [B,H,E,S].
// ---------------------------------------------------------------------------
__global__ __launch_bounds__(256)
void qkv_proj_kernel(const float* __restrict__ q_in, const float* __restrict__ k_in,
                     const float* __restrict__ v_in,
                     const float* __restrict__ Wq, const float* __restrict__ bq,
                     const float* __restrict__ Wk, const float* __restrict__ bk,
                     const float* __restrict__ Wv, const float* __restrict__ bv,
                     _Float16* __restrict__ Qf, _Float16* __restrict__ Kf,
                     _Float16* __restrict__ Vt)
{
    __shared__ __align__(32) _Float16 Wlds[Ex * Ex];       // 32 KB f16 weights
    __shared__ __align__(32) _Float16 Otile[8][16 * Ex];   // 32 KB staging (Q/K)

    const int tid  = threadIdx.x;
    const int lane = tid & 31;
    const int wave = tid >> 5;
    const int l16  = lane & 15;
    const int hi   = lane >> 4;
    const int mat  = blockIdx.y;                            // 0=Q 1=K 2=V

    const float* X; const float* W; const float* bias;
    if (mat == 0)      { X = q_in; W = Wq; bias = bq; }
    else if (mat == 1) { X = k_in; W = Wk; bias = bk; }
    else               { X = v_in; W = Wv; bias = bv; }

    // ---- stage W as f16 in LDS (each thread: 8 runs of 8 contiguous floats, stride 2048) ----
#pragma unroll
    for (int i = 0; i < 8; ++i) {
        const int base = tid * 8 + i * 2048;
        float4 w0 = *(const float4*)(W + base);
        float4 w1 = *(const float4*)(W + base + 4);
        *(v8h*)(Wlds + base) = pack8(w0, w1);
    }
    __syncthreads();

    // ---- tile = (b, h, s0..s0+15) ----
    const int tindex = blockIdx.x * 8 + wave;               // 0..4095
    const int b  = tindex >> 11;
    const int hh = (tindex >> 7) & (Hx - 1);
    const int s0 = (tindex & 127) * 16;
    const int bhE = b * Hx + hh;

    // ---- A fragments: row = input token (s0+l16), 4 e-steps of 32 ----
    const int arow = ((b * Sx + s0 + l16) * Hx + hh) * Ex;
    v16h afrag[4];
#pragma unroll
    for (int es = 0; es < 4; ++es) {
        const float* p = X + arow + es * 32 + hi * 16;
        float4 a0 = *(const float4*)p;
        float4 a1 = *(const float4*)(p + 4);
        float4 a2 = *(const float4*)(p + 8);
        float4 a3 = *(const float4*)(p + 12);
        v8h lo = pack8(a0, a1);
        v8h up = pack8(a2, a3);
#pragma unroll
        for (int j = 0; j < 8; ++j) { afrag[es][j] = lo[j]; afrag[es][j + 8] = up[j]; }
    }

#pragma unroll
    for (int nt = 0; nt < 8; ++nt) {
        const int ncol = nt * 16 + l16;
        const float bval = bias[ncol];
        v8f c;
#pragma unroll
        for (int i = 0; i < 8; ++i) c[i] = bval;
#pragma unroll
        for (int es = 0; es < 4; ++es) {
            v16h bf = *(const v16h*)(Wlds + ncol * Ex + es * 32 + hi * 16);
            c = wmma_f16(afrag[es], bf, c);
        }
        if (mat == 2) {
            // Vt[(bh*E + ncol)*S + s]: 8 consecutive tokens per lane -> one b128
            v8h pk;
#pragma unroll
            for (int r = 0; r < 8; ++r) pk[r] = (_Float16)c[r];
            const int voff = (bhE * Ex + ncol) * Sx + s0 + hi * 8;
            *(v8h*)(Vt + voff) = pk;
        } else {
            _Float16* ot = &Otile[wave][0];
#pragma unroll
            for (int r = 0; r < 8; ++r) ot[(hi * 8 + r) * Ex + ncol] = (_Float16)c[r];
        }
    }

    if (mat != 2) {
        __syncthreads();                                     // order LDS tile writes
        _Float16* O = (mat == 0) ? Qf : Kf;
        const int obase = (bhE * Sx + s0) * Ex;              // 16 rows = 4KB contiguous
#pragma unroll
        for (int i = 0; i < 8; ++i) {
            const int off = (i * 32 + lane) * 8;             // coalesced 512B/instr
            *(v8h*)(O + obase + off) = *(const v8h*)(&Otile[wave][off]);
        }
    }
}

// ---------------------------------------------------------------------------
// Kernel 2: flash attention.  8 waves/block, each owns 16 q-rows.  K/V tiles
// (shared by all 8 waves) are double-buffered in LDS via the Tensor Data
// Mover (wave 0 issues, s_wait_tensorcnt + barrier).  Per kv-tile of 32:
// 8 WMMAs for S=Q*K^T, online softmax in C-layout, P transposed through a
// per-wave LDS slice, 8 WMMAs for O += P*V.
// ---------------------------------------------------------------------------
__global__ __launch_bounds__(256)
void flash_attn_kernel(const _Float16* __restrict__ Qf, const _Float16* __restrict__ Kf,
                       const _Float16* __restrict__ Vt, const float* __restrict__ mask,
                       float* __restrict__ out)
{
    __shared__ __align__(64) _Float16 Ktile[2][32 * Ex];    // 2 x 8 KB  [kk][e]
    __shared__ __align__(64) _Float16 Vtile[2][Ex * 32];    // 2 x 8 KB  [f][kk]
    __shared__ __align__(16) v8h pbuf[8][16][4];            // 8 KB P transpose

    const int lane = threadIdx.x & 31;
    const int wave = threadIdx.x >> 5;
    const int l16  = lane & 15;
    const int hi   = lane >> 4;
    const int bh   = blockIdx.z * Hx + blockIdx.y;
    const int q0   = blockIdx.x * 128 + wave * 16;

    const _Float16* Qb = Qf + bh * Sx * Ex;
    const _Float16* Kb = Kf + bh * Sx * Ex;
    const _Float16* Vb = Vt + bh * Sx * Ex;                 // [E][S] slice

    // ---- Q A-fragments (persistent) ----
    const _Float16* qrow = Qb + (q0 + l16) * Ex;
    v16h aq[4];
#pragma unroll
    for (int es = 0; es < 4; ++es) {
        const int c0 = es * 32 + hi * 16;
        v8h lo = *(const v8h*)(qrow + c0);
        v8h up = *(const v8h*)(qrow + c0 + 8);
#pragma unroll
        for (int j = 0; j < 8; ++j) { aq[es][j] = lo[j]; aq[es][j + 8] = up[j]; }
    }

    float mi[8], li[8];
    v8f acc[8];
#pragma unroll
    for (int r = 0; r < 8; ++r) { mi[r] = -1e30f; li[r] = 0.0f; }
#pragma unroll
    for (int ft = 0; ft < 8; ++ft)
#pragma unroll
        for (int i = 0; i < 8; ++i) acc[ft][i] = 0.0f;

    const float invs = 0.08838834764831845f;                // 1/sqrt(128)
    const float L2E  = 1.44269504088896340f;

#ifdef USE_TDM
    if (wave == 0) {
        tdm_load_2d((unsigned)(uintptr_t)&Ktile[0][0], Kb, Ex, 32, Ex, 32, Ex);
        tdm_load_2d((unsigned)(uintptr_t)&Vtile[0][0], Vb, 32, Ex, 32, Ex, Sx);
    }
#endif
    int cur = 0;

    for (int kv = 0; kv < Sx; kv += 32) {
#ifdef USE_TDM
        if (wave == 0) __builtin_amdgcn_s_wait_tensorcnt(0);
        __syncthreads();                                     // publish buf[cur]
        if (wave == 0 && kv + 32 < Sx) {
            const int nxt = cur ^ 1;
            tdm_load_2d((unsigned)(uintptr_t)&Ktile[nxt][0], Kb + (kv + 32) * Ex,
                        Ex, 32, Ex, 32, Ex);
            tdm_load_2d((unsigned)(uintptr_t)&Vtile[nxt][0], Vb + (kv + 32),
                        32, Ex, 32, Ex, Sx);
        }
#else
        __syncthreads();                                     // WAR on shared tile
        {   // cooperative synchronous copy of this kv tile
            const int t = threadIdx.x;
            *(v16h*)(&Ktile[0][t * 16]) = *(const v16h*)(Kb + kv * Ex + t * 16);
            const int f = t >> 1, hf = t & 1;
            *(v16h*)(&Vtile[0][f * 32 + hf * 16]) =
                *(const v16h*)(Vb + f * Sx + kv + hf * 16);
        }
        __syncthreads();
        cur = 0;
#endif

        // ---- S = Q * K^T ----
        v8f c0v, c1v;
#pragma unroll
        for (int i = 0; i < 8; ++i) { c0v[i] = 0.0f; c1v[i] = 0.0f; }
#pragma unroll
        for (int es = 0; es < 4; ++es) {
            v16h b0 = *(const v16h*)(&Ktile[cur][(l16)      * Ex + es * 32 + hi * 16]);
            v16h b1 = *(const v16h*)(&Ktile[cur][(16 + l16) * Ex + es * 32 + hi * 16]);
            c0v = wmma_f16(aq[es], b0, c0v);
            c1v = wmma_f16(aq[es], b1, c1v);
        }

        // ---- scale + mask (32-bit offsets) ----
        float sc0[8], sc1[8];
        const int qrbase = q0 + hi * 8;
#pragma unroll
        for (int i = 0; i < 8; ++i) {
            const int mrow = (qrbase + i) * Sx + kv;
            sc0[i] = c0v[i] * invs + mask[mrow + l16];
            sc1[i] = c1v[i] * invs + mask[mrow + 16 + l16];
        }

        // ---- online softmax (reductions inside 16-lane halves) ----
        float mnew[8], alpha[8];
#pragma unroll
        for (int i = 0; i < 8; ++i) {
            float t = fmaxf(sc0[i], sc1[i]);
            t = fmaxf(t, __shfl_xor(t, 1, 32));
            t = fmaxf(t, __shfl_xor(t, 2, 32));
            t = fmaxf(t, __shfl_xor(t, 4, 32));
            t = fmaxf(t, __shfl_xor(t, 8, 32));
            mnew[i]  = fmaxf(mi[i], t);
            alpha[i] = exp2f((mi[i] - mnew[i]) * L2E);
        }
#pragma unroll
        for (int i = 0; i < 8; ++i) {
            sc0[i] = exp2f((sc0[i] - mnew[i]) * L2E);
            sc1[i] = exp2f((sc1[i] - mnew[i]) * L2E);
            float t = sc0[i] + sc1[i];
            t += __shfl_xor(t, 1, 32);
            t += __shfl_xor(t, 2, 32);
            t += __shfl_xor(t, 4, 32);
            t += __shfl_xor(t, 8, 32);
            li[i] = li[i] * alpha[i] + t;
            mi[i] = mnew[i];
        }

        // ---- P: C-layout -> A-layout via per-wave LDS slice ----
        {
            _Float16* pr = (_Float16*)&pbuf[wave][0][0];
#pragma unroll
            for (int i = 0; i < 8; ++i) {
                const int m = hi * 8 + i;
                pr[m * 32 + l16]      = (_Float16)sc0[i];
                pr[m * 32 + 16 + l16] = (_Float16)sc1[i];
            }
        }
        __syncthreads();
        v16h pa;
        {
            v8h plo = pbuf[wave][l16][2 * hi];
            v8h phi = pbuf[wave][l16][2 * hi + 1];
#pragma unroll
            for (int j = 0; j < 8; ++j) { pa[j] = plo[j]; pa[j + 8] = phi[j]; }
        }

        // ---- O = diag(alpha)*O + P * V ----
#pragma unroll
        for (int ft = 0; ft < 8; ++ft) {
            v16h bvf = *(const v16h*)(&Vtile[cur][(ft * 16 + l16) * 32 + hi * 16]);
            v8f cacc = acc[ft];
#pragma unroll
            for (int i = 0; i < 8; ++i) cacc[i] *= alpha[i];
            acc[ft] = wmma_f16(pa, bvf, cacc);
        }
#ifdef USE_TDM
        cur ^= 1;
#endif
    }

    // ---- epilogue: normalize and store [B,H,S,E] fp32 ----
    float rli[8];
#pragma unroll
    for (int i = 0; i < 8; ++i) rli[i] = 1.0f / li[i];
    const int obase = (bh * Sx + q0 + hi * 8) * Ex + l16;
#pragma unroll
    for (int ft = 0; ft < 8; ++ft)
#pragma unroll
        for (int i = 0; i < 8; ++i)
            out[obase + i * Ex + ft * 16] = acc[ft][i] * rli[i];
}

extern "C" void kernel_launch(void* const* d_in, const int* in_sizes, int n_in,
                              void* d_out, int out_size, void* d_ws, size_t ws_size,
                              hipStream_t stream)
{
    const float* query = (const float*)d_in[0];
    const float* key   = (const float*)d_in[1];
    const float* value = (const float*)d_in[2];
    const float* mask  = (const float*)d_in[3];
    const float* Wq    = (const float*)d_in[4];
    const float* bq    = (const float*)d_in[5];
    const float* Wk    = (const float*)d_in[6];
    const float* bk    = (const float*)d_in[7];
    const float* Wv    = (const float*)d_in[8];
    const float* bv    = (const float*)d_in[9];

    const long NTOK = (long)Bx * Hx * Sx * Ex;
    _Float16* Qf = (_Float16*)d_ws;
    _Float16* Kf = Qf + NTOK;
    _Float16* Vt = Kf + NTOK;

    qkv_proj_kernel<<<dim3(512, 3, 1), 256, 0, stream>>>(
        query, key, value, Wq, bq, Wk, bk, Wv, bv, Qf, Kf, Vt);

    flash_attn_kernel<<<dim3(Sx / 128, Hx, Bx), 256, 0, stream>>>(
        Qf, Kf, Vt, mask, (float*)d_out);
}
